// GAE_14370960572699
// MI455X (gfx1250) — matrix operand order repaired and checked by reading, not compiled, not run on hardware
//
#include <hip/hip_runtime.h>
#include <hip/hip_bf16.h>

typedef __attribute__((ext_vector_type(16))) _Float16 v16h;
typedef __attribute__((ext_vector_type(8)))  float    v8f;

#define APITCH 40        // A LDS row pitch in halves: 32 + 8 pad (80 B rows)
#define BFRAG_PITCH 24   // B LDS per-lane pitch in halves: 16 + 8 pad (48 B)
#define BSUB_STRIDE (32 * BFRAG_PITCH)   // one 16-col subtile = 32 lanes

// ---------------------------------------------------------------------------
// Degree / normalization kernels
// ---------------------------------------------------------------------------
__global__ void deg_init(float* deg, int n) {
    int i = blockIdx.x * blockDim.x + threadIdx.x;
    if (i < n) deg[i] = 1.0f;  // self-loop contributes 1
}

__global__ void deg_accum(const long long* __restrict__ dst, float* deg, int e) {
    int i = blockIdx.x * blockDim.x + threadIdx.x;
    if (i < e) unsafeAtomicAdd(&deg[(int)dst[i]], 1.0f);
}

__global__ void deg_to_dinv(float* deg, int n) {
    int i = blockIdx.x * blockDim.x + threadIdx.x;
    if (i < n) {
        float d = deg[i];
        deg[i] = (d > 0.0f) ? rsqrtf(d) : 0.0f;
    }
}

// ---------------------------------------------------------------------------
// WMMA GEMM: C[M,N] = A[M,K] * B[K,N], fp32 in/out, f16 WMMA with f32 accum.
// Block tile 128x64, 256 threads = 8 waves, each wave a 32x32 tile (2x2 WMMA).
// B is staged in LDS in WMMA-fragment register layout so each B fragment is
// exactly two aligned ds_load_b128s (no scalar u16 gather).
// ---------------------------------------------------------------------------
__global__ __launch_bounds__(256) void gemm_f32_wmma(
    const float* __restrict__ A, const float* __restrict__ B,
    float* __restrict__ C, int M, int N, int K) {
    __shared__ _Float16 Ah[128 * APITCH];      // 10 240 B
    __shared__ _Float16 Bs[4 * BSUB_STRIDE];   //  6 144 B (4 subtiles of 16 cols)

    const int tid  = threadIdx.x;
    const int lane = tid & 31;
    const int wave = tid >> 5;
    const int wm   = wave & 3;   // wave row 0..3  -> rows wm*32
    const int wn   = wave >> 2;  // wave col 0..1  -> cols wn*32
    const int block_row = blockIdx.x * 128;
    const int block_col = blockIdx.y * 64;

    v8f acc[2][2] = {};

    const int mlane = lane & 15;
    const int hlane = lane >> 4;

    // B staging coordinates: thread t -> column (t&63), K segment (t>>6)*8
    const int bn    = tid & 63;        // 0..63 column within block tile
    const int bkseg = tid >> 6;        // 0..3  -> k = bkseg*8
    const int bsub  = bn >> 4;         // subtile 0..3
    const int blane = ((bkseg & 1) << 4) + (bn & 15);   // ((k/8)&1)*16 + n%16
    const int bj0   = (bkseg >> 1) << 3;                // 0 or 8 halves
    _Float16* bdst = Bs + bsub * BSUB_STRIDE + blane * BFRAG_PITCH + bj0;

    for (int k0 = 0; k0 < K; k0 += 32) {
        // ---- stage A tile (128x32) into LDS as f16: 2 threads per row ----
        {
            int r  = tid >> 1;
            int cb = (tid & 1) * 16;
            int grow = block_row + r;
            _Float16* dp = Ah + r * APITCH + cb;
            if (grow < M) {
                const float* sp = A + (size_t)grow * K + k0 + cb;
#pragma unroll
                for (int j = 0; j < 16; j += 4) {
                    float4 v = *(const float4*)(sp + j);
                    dp[j + 0] = (_Float16)v.x;
                    dp[j + 1] = (_Float16)v.y;
                    dp[j + 2] = (_Float16)v.z;
                    dp[j + 3] = (_Float16)v.w;
                }
            } else {
#pragma unroll
                for (int j = 0; j < 16; ++j) dp[j] = (_Float16)0.0f;
            }
        }
        // ---- stage B tile (32x64) into LDS in fragment layout ----
        // thread reads 8 consecutive-K elements of one column (coalesced b32
        // across threads), packs to 8 halves, one 16B-aligned ds_store_b128.
        {
            const float* sp = B + (size_t)(k0 + bkseg * 8) * N + block_col + bn;
            union { _Float16 h[8]; float4 q; } pk;
#pragma unroll
            for (int j = 0; j < 8; ++j)
                pk.h[j] = (_Float16)sp[(size_t)j * N];
            *(float4*)bdst = pk.q;
        }
        __syncthreads();

        // ---- A fragments (ISA 7.12.2 16-bit A 16x32 layout) ----
        v16h afrag[2], bfrag[2];
#pragma unroll
        for (int t = 0; t < 2; ++t) {
            int rb = wm * 32 + t * 16;
            union { v16h v; unsigned u[8]; } ua;
            const unsigned* rowp =
                (const unsigned*)(Ah + (rb + mlane) * APITCH);
#pragma unroll
            for (int i = 0; i < 8; ++i) {
                int kk = ((i < 4) ? (2 * i) : (16 + 2 * (i - 4))) + hlane * 8;
                ua.u[i] = rowp[kk >> 1];  // packed f16 pair (K even, K odd)
            }
            afrag[t] = ua.v;
        }
        // ---- B fragments: two ds_load_b128 each (pre-swizzled layout) ----
#pragma unroll
        for (int t = 0; t < 2; ++t) {
            int st = wn * 2 + t;  // subtile index 0..3
            const float4* p =
                (const float4*)(Bs + st * BSUB_STRIDE + lane * BFRAG_PITCH);
            union { v16h v; float4 q[2]; } ub;
            ub.q[0] = p[0];
            ub.q[1] = p[1];
            bfrag[t] = ub.v;
        }

        // ---- 2x2 WMMA: D = A x B + C, f32 accumulators ----
#pragma unroll
        for (int i = 0; i < 2; ++i)
#pragma unroll
            for (int j = 0; j < 2; ++j)
                acc[i][j] = __builtin_amdgcn_wmma_f32_16x16x32_f16(
                    false, afrag[i], false, bfrag[j],
                    (short)0, acc[i][j], false, false);
        __syncthreads();
    }

    // ---- store: C/D layout: VGPR v -> M = v + (lane>=16 ? 8 : 0), N = lane&15
#pragma unroll
    for (int i = 0; i < 2; ++i) {
#pragma unroll
        for (int j = 0; j < 2; ++j) {
            int col = block_col + wn * 32 + j * 16 + mlane;
#pragma unroll
            for (int v = 0; v < 8; ++v) {
                int row = block_row + wm * 32 + i * 16 + hlane * 8 + v;
                if (row < M) C[(size_t)row * N + col] = acc[i][j][v];
            }
        }
    }
}

// ---------------------------------------------------------------------------
// out[i,f] = h[i,f]*dinv[i]^2 + b[f]   (self-loop term + bias)
// ---------------------------------------------------------------------------
__global__ void self_init(const float* __restrict__ h, const float* __restrict__ dinv,
                          const float* __restrict__ bias, float* __restrict__ out,
                          int n, int F) {
    size_t i = (size_t)blockIdx.x * blockDim.x + threadIdx.x;
    size_t total = (size_t)n * F;
    if (i >= total) return;
    int node = (int)(i / F);
    int f    = (int)(i % F);
    float di = dinv[node];
    out[i] = h[i] * di * di + bias[f];
}

// ---------------------------------------------------------------------------
// scatter-add: out[dst] += h[src] * dinv[src]*dinv[dst], 4 floats per thread
// (h and out are L2-resident: 51 MB / 26 MB vs 192 MB L2)
// ---------------------------------------------------------------------------
__global__ void scatter_edges(const float* __restrict__ h,
                              const long long* __restrict__ src,
                              const long long* __restrict__ dst,
                              const float* __restrict__ dinv,
                              float* __restrict__ out, int E, int F) {
    int chunks = F >> 2;
    long long idx = (long long)blockIdx.x * blockDim.x + threadIdx.x;
    long long total = (long long)E * chunks;
    if (idx >= total) return;
    int e = (int)(idx / chunks);
    int c = (int)(idx % chunks) * 4;
    int s = (int)src[e];
    int d = (int)dst[e];
    float w = dinv[s] * dinv[d];
    float4 v = *(const float4*)(h + (size_t)s * F + c);
    float* o = out + (size_t)d * F + c;
    unsafeAtomicAdd(o + 0, v.x * w);
    unsafeAtomicAdd(o + 1, v.y * w);
    unsafeAtomicAdd(o + 2, v.z * w);
    unsafeAtomicAdd(o + 3, v.w * w);
}

// ---------------------------------------------------------------------------
// ReLU + deterministic dropout(p=0.2): keep if hash(i) < 0.8*2^32, scale 1.25
// ---------------------------------------------------------------------------
__global__ void relu_dropout(float* __restrict__ g, size_t total) {
    size_t i = (size_t)blockIdx.x * blockDim.x + threadIdx.x;
    if (i >= total) return;
    float v = g[i];
    v = v > 0.0f ? v : 0.0f;
    unsigned x = (unsigned)i * 2654435761u + 0x9e3779b9u;
    x ^= x >> 16; x *= 0x85ebca6bu;
    x ^= x >> 13; x *= 0xc2b2ae35u;
    x ^= x >> 16;
    g[i] = (x < 3435973836u) ? v * 1.25f : 0.0f;  // 0.8 * 2^32 threshold
}

// ---------------------------------------------------------------------------
// softmax over 128 cols, one wave32 per row (4 floats / lane)
// ---------------------------------------------------------------------------
__global__ __launch_bounds__(256) void softmax128(float* __restrict__ out, int n) {
    int wave = (int)(((size_t)blockIdx.x * blockDim.x + threadIdx.x) >> 5);
    int lane = threadIdx.x & 31;
    if (wave >= n) return;
    float* row = out + (size_t)wave * 128;
    float4 v = *(const float4*)(row + lane * 4);
    float m = fmaxf(fmaxf(v.x, v.y), fmaxf(v.z, v.w));
#pragma unroll
    for (int off = 16; off > 0; off >>= 1) m = fmaxf(m, __shfl_xor(m, off, 32));
    v.x = __expf(v.x - m);
    v.y = __expf(v.y - m);
    v.z = __expf(v.z - m);
    v.w = __expf(v.w - m);
    float s = v.x + v.y + v.z + v.w;
#pragma unroll
    for (int off = 16; off > 0; off >>= 1) s += __shfl_xor(s, off, 32);
    float inv = 1.0f / s;
    v.x *= inv; v.y *= inv; v.z *= inv; v.w *= inv;
    *(float4*)(row + lane * 4) = v;
}

// ---------------------------------------------------------------------------
// launch
// ---------------------------------------------------------------------------
extern "C" void kernel_launch(void* const* d_in, const int* in_sizes, int n_in,
                              void* d_out, int out_size, void* d_ws, size_t ws_size,
                              hipStream_t stream) {
    const float*     x   = (const float*)d_in[0];
    const long long* ei  = (const long long*)d_in[1];   // int64 [2, E]
    const float*     W1  = (const float*)d_in[2];
    const float*     b1  = (const float*)d_in[3];
    const float*     W2  = (const float*)d_in[4];
    const float*     b2  = (const float*)d_in[5];

    const int Nn = in_sizes[0] / 512;   // 50000
    const int E  = in_sizes[1] / 2;     // 800000
    const long long* src = ei;
    const long long* dst = ei + E;

    // workspace carve-up (~128.2 MB)
    float* ws   = (float*)d_ws;
    size_t Nal  = ((size_t)Nn + 255) & ~(size_t)255;
    float* dinv = ws;                         // Nal
    float* h1   = dinv + Nal;                 // Nn*256
    float* g1   = h1 + (size_t)Nn * 256;      // Nn*256
    float* h2   = g1 + (size_t)Nn * 256;      // Nn*128
    float* out  = (float*)d_out;              // Nn*128

    // ---- degrees -> dinv (in place) ----
    deg_init<<<(Nn + 255) / 256, 256, 0, stream>>>(dinv, Nn);
    deg_accum<<<(E + 255) / 256, 256, 0, stream>>>(dst, dinv, E);
    deg_to_dinv<<<(Nn + 255) / 256, 256, 0, stream>>>(dinv, Nn);

    // ---- layer 1: h1 = x @ W1 (WMMA) ----
    {
        dim3 grid((Nn + 127) / 128, 256 / 64);
        gemm_f32_wmma<<<grid, 256, 0, stream>>>(x, W1, h1, Nn, 256, 512);
    }
    size_t tot1 = (size_t)Nn * 256;
    self_init<<<(unsigned)((tot1 + 255) / 256), 256, 0, stream>>>(h1, dinv, b1, g1, Nn, 256);
    {
        long long work = (long long)E * 64;  // E * (256/4)
        scatter_edges<<<(unsigned)((work + 255) / 256), 256, 0, stream>>>(
            h1, src, dst, dinv, g1, E, 256);
    }
    relu_dropout<<<(unsigned)((tot1 + 255) / 256), 256, 0, stream>>>(g1, tot1);

    // ---- layer 2: h2 = g1 @ W2 (WMMA) ----
    {
        dim3 grid((Nn + 127) / 128, 128 / 64);
        gemm_f32_wmma<<<grid, 256, 0, stream>>>(g1, W2, h2, Nn, 128, 256);
    }
    size_t tot2 = (size_t)Nn * 128;
    self_init<<<(unsigned)((tot2 + 255) / 256), 256, 0, stream>>>(h2, dinv, b2, out, Nn, 128);
    {
        long long work = (long long)E * 32;  // E * (128/4)
        scatter_edges<<<(unsigned)((work + 255) / 256), 256, 0, stream>>>(
            h2, src, dst, dinv, out, E, 128);
    }

    // ---- softmax rows of 128 ----
    softmax128<<<(unsigned)(((size_t)Nn * 32 + 255) / 256), 256, 0, stream>>>(out, Nn);
}
